// MultiHeadAttentionBlock_46608985096586
// MI455X (gfx1250) — compile-verified, matrix operand order checked by reading, not run on hardware
//
#include <hip/hip_runtime.h>

#define DIM     4096
#define NQH     32
#define NKVH    8
#define HD      128
#define MAXSEQ  4096
#define BATCH   8
#define NSPLIT  8
#define CHUNK   (MAXSEQ / NSPLIT)   // 512 positions per split
#define WCHUNK  (CHUNK / 8)         // 64 positions per wave (8 waves/block)

typedef _Float16 v16h __attribute__((ext_vector_type(16)));
typedef float    v8f  __attribute__((ext_vector_type(8)));

// CDNA5 async global->LDS copy path (device pass only, feature-guarded so the
// kernel still compiles if this toolchain lacks the builtins).
#if defined(__gfx1250__) && __has_builtin(__builtin_amdgcn_global_load_async_to_lds_b128) && __has_builtin(__builtin_amdgcn_s_wait_asynccnt)
#define HAVE_ASYNC_LDS 1
#else
#define HAVE_ASYNC_LDS 0
#endif

#if HAVE_ASYNC_LDS
// Parameter types per hipcc diagnostic: 16-byte int vector pointees,
// arg0 in the global (AS1) space, arg1 in LDS (AS3).
typedef int b128_t __attribute__((vector_size(16)));
typedef __attribute__((address_space(1))) b128_t* gptr128;
typedef __attribute__((address_space(3))) b128_t* lptr128;

// Stage one 32x16-f32 B tile (4 async b128 issues, coalesced 64B rows).
// src = W + row(srow)*N + n0 + scol for this lane; dst likewise lane-offset.
__device__ __forceinline__ void stage_tile(const float* src, float* dst,
                                           size_t rs8 /* 8*N floats */) {
#pragma unroll
  for (int s = 0; s < 4; ++s) {
    __builtin_amdgcn_global_load_async_to_lds_b128(
        (gptr128)(src + (size_t)s * rs8),
        (lptr128)(dst + s * 8 * 16),
        0, 0);
  }
}
#endif

// ---------------------------------------------------------------------------
// C(8 x N) = A(8 x K, zero-padded to 16 rows) * W(K x N), fp32 in/out,
// f16 multiplies via v_wmma_f32_16x16x32_f16. One wave owns one 16-col tile.
// A 16-bit A-tile lane layout: lane L (M = L&15), kb = (L<16)?0:8,
//   element e<8  -> K = kb + e        (8 contiguous floats)
//   element e>=8 -> K = 16 + kb + e-8 (8 contiguous floats)
// B-tile mirrors it with N = lane&15.
// C/D layout: VGPR i, lanes 0-15 -> M=i, N=lane (M=i+8 for lanes 16-31).
//
// B tiles are double-buffered in LDS via global_load_async_to_lds_b128:
// tile i+1's 4 issues go out before s_wait_asynccnt(4), which (async loads
// complete in order) guarantees only tile i is resident -- DMA of the next
// tile overlaps convert+WMMA of the current one.
// ---------------------------------------------------------------------------
__global__ void gemm8_wmma(const float* __restrict__ A, int lda,
                           const float* __restrict__ W, int K, int N,
                           float* __restrict__ C, int ldc) {
#if HAVE_ASYNC_LDS
  __shared__ float ldsB[4][2][32 * 16];       // 2KB tile x 2 buffers per wave
#endif
  const int wave = threadIdx.x >> 5;
  const int lane = threadIdx.x & 31;
  const int tile = blockIdx.x * 4 + wave;
  const int n0   = tile * 16;
  const int r    = lane & 15;
  const int kb   = (lane < 16) ? 0 : 8;
  const bool arow = (r < 8);                  // batches live in rows 0..7
  const float* Ar = A + (size_t)(r & 7) * lda;  // always-valid row; rows 8..15 masked

  v8f acc = {0.f, 0.f, 0.f, 0.f, 0.f, 0.f, 0.f, 0.f};

#if HAVE_ASYNC_LDS
  const int srow = lane >> 2;                 // 0..7   (staging row within group)
  const int scol = (lane & 3) * 4;            // 0,4,8,12 (float4 column)
  const size_t rs8  = (size_t)8 * N;          // 8 W-rows, in floats
  const size_t rs32 = (size_t)32 * N;         // one k-step, in floats
  const float* wsrc = W + (size_t)srow * N + n0 + scol;   // lane's tile-0 source
  float* lbuf0 = &ldsB[wave][0][srow * 16 + scol];        // lane's dst, buffer 0
  float* lbuf1 = &ldsB[wave][1][srow * 16 + scol];        // lane's dst, buffer 1
  float* base0 = ldsB[wave][0];
  float* base1 = ldsB[wave][1];
  stage_tile(wsrc, lbuf0, rs8);               // prologue: tile 0 -> buffer 0
  int buf = 0;
#endif

  for (int k0 = 0; k0 < K; k0 += 32) {
#if HAVE_ASYNC_LDS
    const bool has_next = (k0 + 32 < K);
    if (has_next) {                           // issue tile i+1 into other buffer
      stage_tile(wsrc + rs32, (buf ? lbuf0 : lbuf1), rs8);
      __builtin_amdgcn_s_wait_asynccnt(4);    // only tile i guaranteed resident
    } else {
      __builtin_amdgcn_s_wait_asynccnt(0);
    }
    wsrc += rs32;
#endif
    if (k0 + 64 < K) __builtin_prefetch(Ar + k0 + 64, 0, 3);

    // A fragment: unconditional vector loads + value select (no EXEC branches)
    const float4 a0 = *(const float4*)(Ar + k0 + kb);
    const float4 a1 = *(const float4*)(Ar + k0 + kb + 4);
    const float4 a2 = *(const float4*)(Ar + k0 + kb + 16);
    const float4 a3 = *(const float4*)(Ar + k0 + kb + 20);
    float af[16];
    af[0]  = a0.x; af[1]  = a0.y; af[2]  = a0.z; af[3]  = a0.w;
    af[4]  = a1.x; af[5]  = a1.y; af[6]  = a1.z; af[7]  = a1.w;
    af[8]  = a2.x; af[9]  = a2.y; af[10] = a2.z; af[11] = a2.w;
    af[12] = a3.x; af[13] = a3.y; af[14] = a3.z; af[15] = a3.w;
    v16h a;
#pragma unroll
    for (int e = 0; e < 16; ++e)
      a[e] = (_Float16)(arow ? af[e] : 0.0f);

    v16h b;
#if HAVE_ASYNC_LDS
    const float* bt = buf ? base1 : base0;
#pragma unroll
    for (int e = 0; e < 16; ++e) {
      const int kkl = kb + e + ((e < 8) ? 0 : 8);
      b[e] = (_Float16)bt[kkl * 16 + r];
    }
    buf ^= 1;
#else
#pragma unroll
    for (int e = 0; e < 16; ++e) {
      const int kk = k0 + kb + e + ((e < 8) ? 0 : 8);
      b[e] = (_Float16)W[(size_t)kk * N + n0 + r];
    }
#endif
    acc = __builtin_amdgcn_wmma_f32_16x16x32_f16(false, a, false, b,
                                                 (short)0, acc, false, false);
  }

  if (lane < 16) {
#pragma unroll
    for (int i = 0; i < 8; ++i)               // M = i = batch index
      C[(size_t)i * ldc + n0 + r] = acc[i];
  }
}

// ---------------------------------------------------------------------------
// Interleaved-pair RoPE, in place, on q (8x32x128) and new k (8x8x128).
// freqs_cos/sin are (1 x 64) for the single position start_pos = 4095.
// ---------------------------------------------------------------------------
__global__ void rope_kernel(float* __restrict__ q, float* __restrict__ k,
                            const float* __restrict__ fc,
                            const float* __restrict__ fs) {
  const int NQP = BATCH * NQH * (HD / 2);     // 16384 pairs in q
  const int NKP = BATCH * NKVH * (HD / 2);    // 4096 pairs in k
  const int i = blockIdx.x * blockDim.x + threadIdx.x;
  if (i >= NQP + NKP) return;
  float* buf;
  int p, vec;
  if (i < NQP) { buf = q; p = i & 63; vec = i >> 6; }
  else         { const int j = i - NQP; buf = k; p = j & 63; vec = j >> 6; }
  const float c = fc[p], s = fs[p];
  float* v = buf + (size_t)vec * HD + p * 2;
  const float xr = v[0], xi = v[1];
  v[0] = xr * c - xi * s;
  v[1] = xr * s + xi * c;
}

// ---------------------------------------------------------------------------
// Split-K flash decode. Grid (split, kv_head, batch), 256 threads = 8 waves.
// Each wave streams 64 key positions; one position per step: 32 lanes x
// float4 = the whole 128-dim K/V row, coalesced, reused for all 4 q-heads
// of this kv group. The new token (t = 4095) is read from scratch k/v so
// the cache inputs are never mutated. Emits unnormalized partial (O, m, l).
// ---------------------------------------------------------------------------
__global__ void attn_partial(const float* __restrict__ q,
                             const float* __restrict__ krope,
                             const float* __restrict__ vnew,
                             const float* __restrict__ ck,
                             const float* __restrict__ cv,
                             float* __restrict__ po,
                             float* __restrict__ pml) {
  const int sp = blockIdx.x, kv = blockIdx.y, b = blockIdx.z;
  const int wave = threadIdx.x >> 5, lane = threadIdx.x & 31;
  const float scale = 0.08838834764831845f;   // 1/sqrt(128)

  __shared__ float sm[8][4];
  __shared__ float sl[8][4];
  __shared__ float sacc[8][4][HD];

  const float* qb = q + ((size_t)b * NQH + kv * 4) * HD;
  float4 qf[4];
#pragma unroll
  for (int j = 0; j < 4; ++j)
    qf[j] = *(const float4*)(qb + j * HD + lane * 4);

  float m[4], l[4];
  float4 acc[4];
#pragma unroll
  for (int j = 0; j < 4; ++j) {
    m[j] = -1e30f; l[j] = 0.f; acc[j] = make_float4(0.f, 0.f, 0.f, 0.f);
  }

  const int t0 = sp * CHUNK + wave * WCHUNK;
  for (int tt = 0; tt < WCHUNK; ++tt) {
    const int t = t0 + tt;
    const float *kp, *vp;
    if (t == MAXSEQ - 1) {                    // the freshly computed token
      kp = krope + ((size_t)b * NKVH + kv) * HD;
      vp = vnew  + ((size_t)b * NKVH + kv) * HD;
    } else {
      const size_t off = (((size_t)b * MAXSEQ + t) * NKVH + kv) * HD;
      kp = ck + off; vp = cv + off;
    }
    const float4 kvv = *(const float4*)(kp + lane * 4);
    const float4 vvv = *(const float4*)(vp + lane * 4);
#pragma unroll
    for (int j = 0; j < 4; ++j) {
      float d = kvv.x * qf[j].x + kvv.y * qf[j].y +
                kvv.z * qf[j].z + kvv.w * qf[j].w;
      d += __shfl_xor(d, 16); d += __shfl_xor(d, 8); d += __shfl_xor(d, 4);
      d += __shfl_xor(d, 2);  d += __shfl_xor(d, 1);
      const float s  = d * scale;
      const float mn = fmaxf(m[j], s);
      const float f  = __expf(m[j] - mn);
      const float p  = __expf(s - mn);
      l[j] = l[j] * f + p;
      acc[j].x = acc[j].x * f + p * vvv.x;
      acc[j].y = acc[j].y * f + p * vvv.y;
      acc[j].z = acc[j].z * f + p * vvv.z;
      acc[j].w = acc[j].w * f + p * vvv.w;
      m[j] = mn;
    }
  }

#pragma unroll
  for (int j = 0; j < 4; ++j) {
    sacc[wave][j][lane * 4 + 0] = acc[j].x;
    sacc[wave][j][lane * 4 + 1] = acc[j].y;
    sacc[wave][j][lane * 4 + 2] = acc[j].z;
    sacc[wave][j][lane * 4 + 3] = acc[j].w;
    if (lane == 0) { sm[wave][j] = m[j]; sl[wave][j] = l[j]; }
  }
  __syncthreads();

  const int d  = threadIdx.x & 127;
  const int jh = threadIdx.x >> 7;            // threads 0-127 -> j {0,1}, 128-255 -> j {2,3}
#pragma unroll
  for (int jj = 0; jj < 2; ++jj) {
    const int j = jh * 2 + jj;
    float M = -1e30f;
#pragma unroll
    for (int w = 0; w < 8; ++w) M = fmaxf(M, sm[w][j]);
    float L = 0.f, O = 0.f;
#pragma unroll
    for (int w = 0; w < 8; ++w) {
      const float f = __expf(sm[w][j] - M);
      L += sl[w][j] * f;
      O += sacc[w][j][d] * f;
    }
    const int h = kv * 4 + j;
    const size_t base = (size_t)(b * NQH + h) * NSPLIT + sp;
    po[base * HD + d] = O;
    if (d == 0) { pml[base * 2] = M; pml[base * 2 + 1] = L; }
  }
}

// ---------------------------------------------------------------------------
// LSE-merge the NSPLIT partials and normalize -> attn_out (b, h*128+d).
// ---------------------------------------------------------------------------
__global__ void attn_combine(const float* __restrict__ po,
                             const float* __restrict__ pml,
                             float* __restrict__ ao) {
  const int bh = blockIdx.x;                  // b*32 + h
  const int d  = threadIdx.x;                 // 0..127
  float M = -1e30f;
#pragma unroll
  for (int sp = 0; sp < NSPLIT; ++sp)
    M = fmaxf(M, pml[((size_t)bh * NSPLIT + sp) * 2]);
  float L = 0.f, O = 0.f;
#pragma unroll
  for (int sp = 0; sp < NSPLIT; ++sp) {
    const float mm = pml[((size_t)bh * NSPLIT + sp) * 2];
    const float ll = pml[((size_t)bh * NSPLIT + sp) * 2 + 1];
    const float f  = __expf(mm - M);
    L += ll * f;
    O += po[((size_t)bh * NSPLIT + sp) * HD + d] * f;
  }
  ao[(size_t)bh * HD + d] = O / L;            // bh*128 == b*4096 + h*128
}

// ---------------------------------------------------------------------------
extern "C" void kernel_launch(void* const* d_in, const int* in_sizes, int n_in,
                              void* d_out, int out_size, void* d_ws, size_t ws_size,
                              hipStream_t stream) {
  const float* x  = (const float*)d_in[0];
  // d_in[1] = start_pos (4095, fixed by the reference constants)
  const float* fc = (const float*)d_in[2];
  const float* fs = (const float*)d_in[3];
  const float* wq = (const float*)d_in[4];
  const float* wk = (const float*)d_in[5];
  const float* wv = (const float*)d_in[6];
  const float* wo = (const float*)d_in[7];
  const float* ck = (const float*)d_in[8];
  const float* cv = (const float*)d_in[9];
  float* out = (float*)d_out;

  // workspace carve-up (floats): 348,160 floats ~= 1.4 MB
  float* q_raw = (float*)d_ws;                        // 8*4096
  float* k_raw = q_raw + BATCH * DIM;                 // 8*8*128
  float* v_new = k_raw + BATCH * NKVH * HD;           // 8*8*128
  float* ao    = v_new + BATCH * NKVH * HD;           // 8*4096
  float* po    = ao + BATCH * DIM;                    // 8*32*8*128
  float* pml   = po + (size_t)BATCH * NQH * NSPLIT * HD; // 8*32*8*2

  // q/k/v projections (WMMA GEMV, M=8 padded to 16)
  gemm8_wmma<<<DIM / 64, 128, 0, stream>>>(x, DIM, wq, DIM, DIM, q_raw, DIM);
  gemm8_wmma<<<(NKVH * HD) / 64, 128, 0, stream>>>(x, DIM, wk, DIM, NKVH * HD,
                                                   k_raw, NKVH * HD);
  gemm8_wmma<<<(NKVH * HD) / 64, 128, 0, stream>>>(x, DIM, wv, DIM, NKVH * HD,
                                                   v_new, NKVH * HD);
  // RoPE on q and the new k
  rope_kernel<<<80, 256, 0, stream>>>(q_raw, k_raw, fc, fs);
  // split-K flash decode over the 4096-entry cache (+ new token at t=4095)
  attn_partial<<<dim3(NSPLIT, NKVH, BATCH), 256, 0, stream>>>(
      q_raw, k_raw, v_new, ck, cv, po, pml);
  attn_combine<<<BATCH * NQH, HD, 0, stream>>>(po, pml, ao);
  // output projection
  gemm8_wmma<<<DIM / 64, 128, 0, stream>>>(ao, DIM, wo, DIM, DIM, out, DIM);
}
